// TGNN_41832981463293
// MI455X (gfx1250) — matrix operand order repaired and checked by reading, not compiled, not run on hardware
//
#include <hip/hip_runtime.h>
#include <stdint.h>

// ---------------- problem constants ----------------
#define NN    2048
#define TT    64
#define EE    16384
#define BB    64
#define K0    32
#define EMB   128
#define K1    64
#define K2    64
#define KS    7
#define POOL  4
#define EPS   1e-5f
#define MROWS (NN*TT)          // 131072

// ---------------- workspace layout (bytes) ----------------
#define OFF_DEG    0u                      // N*4        = 8192
#define OFF_NORM   8192u                   // E*4        = 65536
#define OFF_SC1    73728u                  // 32*4
#define OFF_SH1    73856u
#define OFF_SC2    73984u                  // 64*4
#define OFF_SH2    74240u
#define OFF_SC3    74496u
#define OFF_SH3    74752u
#define OFF_WPK1   75008u                  // 2048 u32   (gcn1_w packed)
#define OFF_WPK2   83200u                  // 8192 u32   (gcn2_w packed)
#define OFF_WPK2C  115968u                 // 28672 u32  (conv2_w packed)
#define OFF_H0     230656u                 // N*T*32 bf16  = 8 MB
#define OFF_H      8619264u                // N*T*128 bf16 = 32 MB
#define OFF_Y      42173696u               // N*T*128 bf16 = 32 MB
#define OFF_ACC    75728128u               // N*T*128 f32  = 64 MB
#define OFF_P      142836992u              // N*14*64 f32
#define OFF_G      150177024u              // B*14*64 f32

typedef __attribute__((ext_vector_type(16))) __bf16 v16bf;
typedef __attribute__((ext_vector_type(8)))  float  v8f;

union FragU { uint4 q[2]; uint32_t u[8]; v16bf v; };

static __device__ __forceinline__ uint32_t f2bf(float f) {
    uint32_t u = __float_as_uint(f);
    return (u + 0x7FFFu + ((u >> 16) & 1u)) >> 16;   // RNE
}
static __device__ __forceinline__ float bf2f(uint32_t h) {
    return __uint_as_float(h << 16);
}

// ---------------- BN scale/shift precompute ----------------
__global__ void k_prep(const float* g1, const float* b1, const float* m1, const float* v1,
                       const float* g2, const float* b2, const float* m2, const float* v2,
                       const float* g3, const float* b3, const float* m3, const float* v3,
                       float* sc1, float* sh1, float* sc2, float* sh2, float* sc3, float* sh3) {
    int t = threadIdx.x;
    if (t < 32) { float s = g1[t] * rsqrtf(v1[t] + EPS); sc1[t] = s; sh1[t] = b1[t] - m1[t] * s; }
    if (t < 64) { float s = g2[t] * rsqrtf(v2[t] + EPS); sc2[t] = s; sh2[t] = b2[t] - m2[t] * s; }
    if (t < 64) { float s = g3[t] * rsqrtf(v3[t] + EPS); sc3[t] = s; sh3[t] = b3[t] - m3[t] * s; }
}

// ---------------- pack W (K,NC) fp32 -> WMMA B-fragment order, bf16 ----------
// dst layout: [tap][kc][ct][lane][j]  (j = 8 dwords = VGPR index)
// lane: n = ct*16 + (lane&15), khalf = lane>>4 ; VGPR j holds K = kc*32 + (j>>2)*16 + khalf*8 + 2*(j&3), +1
__global__ void k_packW(const float* __restrict__ W, uint32_t* __restrict__ dst,
                        int taps, int KC, int NCT, int NC, int tapStrideW) {
    int tid = blockIdx.x * blockDim.x + threadIdx.x;
    int total = taps * KC * NCT * 32;
    if (tid >= total) return;
    int per  = KC * NCT * 32;
    int tap  = tid / per;
    int r    = tid % per;
    int kc   = r / (NCT * 32);
    int r2   = r % (NCT * 32);
    int ct   = r2 / 32;
    int lane = r2 % 32;
    int hi   = lane >> 4;
    int n    = ct * 16 + (lane & 15);
    const float* Ws = W + (size_t)tap * tapStrideW;
    uint32_t* dl = dst + (size_t)tid * 8;
#pragma unroll
    for (int j = 0; j < 8; j++) {
        int k0 = kc * 32 + (j >> 2) * 16 + hi * 8 + 2 * (j & 3);
        uint32_t lo = f2bf(Ws[(size_t)k0 * NC + n]);
        uint32_t hb = f2bf(Ws[(size_t)(k0 + 1) * NC + n]);
        dl[j] = lo | (hb << 16);
    }
}

// ---------------- degree / norm ----------------
__global__ void k_deginit(float* deg) {
    int i = blockIdx.x * blockDim.x + threadIdx.x;
    if (i < NN) deg[i] = 1.0f;                       // self-loop
}
__global__ void k_degacc(const int* __restrict__ dst, float* deg) {
    int e = blockIdx.x * blockDim.x + threadIdx.x;
    if (e < EE) atomicAdd(&deg[dst[e]], 1.0f);
}
__global__ void k_norm(const int* __restrict__ src, const int* __restrict__ dst,
                       const float* __restrict__ deg, float* norm) {
    int e = blockIdx.x * blockDim.x + threadIdx.x;
    if (e < EE) norm[e] = rsqrtf(deg[src[e]]) * rsqrtf(deg[dst[e]]);
}

// ---------------- conv1 + bn1 + relu -> bf16 (N,T,32) ----------------
__global__ void k_conv1(const float* __restrict__ x, const float* __restrict__ w1,
                        const float* __restrict__ sc1, const float* __restrict__ sh1,
                        uint16_t* __restrict__ h0) {
    int g = blockIdx.x * blockDim.x + threadIdx.x;
    if (g >= MROWS) return;
    int n = g >> 6, t = g & 63;
    float xw[KS];
#pragma unroll
    for (int j = 0; j < KS; j++) {
        int tt = t - 3 + j;
        xw[j] = (tt >= 0 && tt < TT) ? x[(size_t)n * TT + tt] : 0.0f;
    }
    uint32_t out[16];
#pragma unroll
    for (int c2 = 0; c2 < 16; c2++) {
        float a0 = 0.f, a1 = 0.f;
#pragma unroll
        for (int j = 0; j < KS; j++) {
            a0 = fmaf(xw[j], w1[j * K0 + 2 * c2],     a0);
            a1 = fmaf(xw[j], w1[j * K0 + 2 * c2 + 1], a1);
        }
        float v0 = fmaxf(fmaf(sc1[2 * c2],     a0, sh1[2 * c2]),     0.f);
        float v1 = fmaxf(fmaf(sc1[2 * c2 + 1], a1, sh1[2 * c2 + 1]), 0.f);
        out[c2] = f2bf(v0) | (f2bf(v1) << 16);
    }
    uint4* dst = (uint4*)(h0 + (size_t)g * K0);
#pragma unroll
    for (int q = 0; q < 4; q++)
        dst[q] = make_uint4(out[4 * q], out[4 * q + 1], out[4 * q + 2], out[4 * q + 3]);
}

// ---------------- WMMA GEMM: Y(bf16, Mx128) = A(bf16, M x KCH*32) @ Wpk ----------
template <int KCH>
__global__ void __launch_bounds__(256)
k_gemm(const uint16_t* __restrict__ A, const uint32_t* __restrict__ Wpk,
       uint16_t* __restrict__ Y) {
    const int lane = threadIdx.x & 31;
    const int wave = threadIdx.x >> 5;
    const int m    = lane & 15;
    const int hi   = lane >> 4;
    const size_t row0 = (size_t)blockIdx.x * 128 + wave * 16;
    const int LDK = KCH * 32;

    const uint32_t* arow = (const uint32_t*)(A + (row0 + m) * LDK);
    FragU afr[KCH];
#pragma unroll
    for (int kc = 0; kc < KCH; kc++) {
        afr[kc].q[0] = *(const uint4*)(arow + kc * 16 + hi * 4);
        afr[kc].q[1] = *(const uint4*)(arow + kc * 16 + 8 + hi * 4);
    }
    __builtin_prefetch(A + (row0 + 128) * LDK, 0, 1);   // global_prefetch_b8

#pragma unroll
    for (int ct = 0; ct < 8; ct++) {
        v8f acc = {};
#pragma unroll
        for (int kc = 0; kc < KCH; kc++) {
            FragU bfr;
            const uint32_t* bp = Wpk + ((size_t)(kc * 8 + ct) * 32 + lane) * 8;
            bfr.q[0] = *(const uint4*)bp;
            bfr.q[1] = *(const uint4*)(bp + 4);
            acc = __builtin_amdgcn_wmma_f32_16x16x32_bf16(
                false, afr[kc].v, false, bfr.v, (short)0, acc, false, false);
        }
#pragma unroll
        for (int r = 0; r < 8; r++)
            Y[(row0 + r + 8 * hi) * 128 + ct * 16 + m] = (uint16_t)f2bf(acc[r]);
    }
}

// ---------------- scatter: ACC init with self-loop term ----------------
__global__ void k_accinit(const uint16_t* __restrict__ Y, const float* __restrict__ deg,
                          float* __restrict__ ACC) {
    int tid = blockIdx.x * blockDim.x + threadIdx.x;       // N*T*16
    int nt = tid >> 4, seg = tid & 15;
    int n = nt >> 6;
    float inv = 1.0f / deg[n];
    const uint4 yv = *(const uint4*)(Y + (size_t)nt * 128 + seg * 8);
    float* o = ACC + (size_t)nt * 128 + seg * 8;
    uint32_t uu[4] = {yv.x, yv.y, yv.z, yv.w};
#pragma unroll
    for (int i = 0; i < 4; i++) {
        o[2 * i]     = bf2f(uu[i] & 0xFFFFu) * inv;
        o[2 * i + 1] = bf2f(uu[i] >> 16) * inv;
    }
}

// ---------------- scatter: edge messages (fp32 atomics, L2 resident) --------
__global__ void k_scatter(const uint16_t* __restrict__ Y, const int* __restrict__ src,
                          const int* __restrict__ dst, const float* __restrict__ norm,
                          float* __restrict__ ACC) {
    int tid = blockIdx.x * blockDim.x + threadIdx.x;       // E*T*16
    int e = tid >> 10;                                     // T*16 = 1024
    int rem = tid & 1023;
    int t = rem >> 4, seg = rem & 15;
    int s = src[e], d = dst[e];
    float w = norm[e];
    const uint4 yv = *(const uint4*)(Y + ((size_t)s * TT + t) * 128 + seg * 8);
    float* accp = ACC + ((size_t)d * TT + t) * 128 + seg * 8;
    uint32_t uu[4] = {yv.x, yv.y, yv.z, yv.w};
#pragma unroll
    for (int i = 0; i < 4; i++) {
        atomicAdd(accp + 2 * i,     bf2f(uu[i] & 0xFFFFu) * w);
        atomicAdd(accp + 2 * i + 1, bf2f(uu[i] >> 16) * w);
    }
}

// ---------------- bias + relu + cvt -> bf16 H ----------------
__global__ void k_biasrelu(const float* __restrict__ ACC, const float* __restrict__ bias,
                           uint16_t* __restrict__ H) {
    int tid = blockIdx.x * blockDim.x + threadIdx.x;       // N*T*16
    int nt = tid >> 4, seg = tid & 15;
    int c0 = seg * 8;
    const float* a = ACC + (size_t)nt * 128 + c0;
    uint32_t out[4];
#pragma unroll
    for (int i = 0; i < 4; i++) {
        float v0 = fmaxf(a[2 * i]     + bias[c0 + 2 * i],     0.f);
        float v1 = fmaxf(a[2 * i + 1] + bias[c0 + 2 * i + 1], 0.f);
        out[i] = f2bf(v0) | (f2bf(v1) << 16);
    }
    *(uint4*)(H + (size_t)nt * 128 + c0) = make_uint4(out[0], out[1], out[2], out[3]);
}

// ---------------- conv2 as 7 tap-GEMMs (WMMA) + bn2 + avgpool4 + relu ------
__global__ void __launch_bounds__(256)
k_conv2(const uint16_t* __restrict__ H, const uint32_t* __restrict__ Wpk2c,
        const float* __restrict__ sc2, const float* __restrict__ sh2,
        float* __restrict__ P) {
    const int n    = blockIdx.x;
    const int lane = threadIdx.x & 31;
    const int wave = threadIdx.x >> 5;
    const int m    = lane & 15;
    const int hi   = lane >> 4;

    for (int s = 0; s < 2; s++) {
        int tile = wave + s * 8;            // 16 tiles: 4 row-tiles x 4 col-tiles
        int rt = tile >> 2, ct = tile & 3;
        int t0 = rt * 16;
        v8f acc = {};
#pragma unroll
        for (int tap = 0; tap < KS; tap++) {
            int tsrc = t0 + m + tap; if (tsrc > 63) tsrc = 63;     // clamped, unused rows
            const uint32_t* arow = (const uint32_t*)(H + ((size_t)n * TT + tsrc) * 128);
#pragma unroll
            for (int kc = 0; kc < 4; kc++) {
                FragU af, bf_;
                af.q[0] = *(const uint4*)(arow + kc * 16 + hi * 4);
                af.q[1] = *(const uint4*)(arow + kc * 16 + 8 + hi * 4);
                const uint32_t* bp = Wpk2c + ((size_t)((tap * 4 + kc) * 4 + ct) * 32 + lane) * 8;
                bf_.q[0] = *(const uint4*)bp;
                bf_.q[1] = *(const uint4*)(bp + 4);
                acc = __builtin_amdgcn_wmma_f32_16x16x32_bf16(
                    false, af.v, false, bf_.v, (short)0, acc, false, false);
            }
        }
        int c = ct * 16 + m;
        float sc = sc2[c], sh = sh2[c];
        float b0 = fmaf(sc, acc[0], sh), b1 = fmaf(sc, acc[1], sh);
        float b2 = fmaf(sc, acc[2], sh), b3 = fmaf(sc, acc[3], sh);
        float b4 = fmaf(sc, acc[4], sh), b5 = fmaf(sc, acc[5], sh);
        float b6 = fmaf(sc, acc[6], sh), b7 = fmaf(sc, acc[7], sh);
        float p0 = (b0 + b1 + b2 + b3) * 0.25f;
        float p1 = (b4 + b5 + b6 + b7) * 0.25f;
        int tp = t0 / 4 + hi * 2;            // lane rows {r, r+8} -> pool groups
        if (tp < 14)     P[((size_t)n * 14 + tp) * K1 + c]     = fmaxf(p0, 0.f);
        if (tp + 1 < 14) P[((size_t)n * 14 + tp + 1) * K1 + c] = fmaxf(p1, 0.f);
    }
}

// ---------------- per-graph mean pool: (N,14,64) -> (B,14,64) ----------------
__global__ void k_graphpool(const float* __restrict__ P, float* __restrict__ G) {
    int tid = blockIdx.x * blockDim.x + threadIdx.x;       // B*14*64
    if (tid >= BB * 14 * K1) return;
    int b = tid / (14 * K1);
    int rem = tid % (14 * K1);
    float sum = 0.f;
    for (int i = 0; i < NN / BB; i++)
        sum += P[((size_t)(b * (NN / BB) + i) * 14) * K1 + rem];
    G[tid] = sum * (1.0f / (NN / BB));
}

// ---------------- head: conv3+bn3+pool+relu -> dense -> log_softmax ---------
__global__ void k_head(const float* __restrict__ G, const float* __restrict__ w3,
                       const float* __restrict__ sc3, const float* __restrict__ sh3,
                       const float* __restrict__ dw, const float* __restrict__ db,
                       float* __restrict__ out) {
    __shared__ float gs[14 * K1];
    __shared__ float flat[2 * K2];
    __shared__ float logits[4];
    int b = blockIdx.x, c = threadIdx.x;    // 64 threads
    for (int i = c; i < 14 * K1; i += 64) gs[i] = G[(size_t)b * 14 * K1 + i];
    __syncthreads();
    float o[8];
#pragma unroll
    for (int t3 = 0; t3 < 8; t3++) {
        float a = 0.f;
        for (int k = 0; k < KS; k++)
            for (int ci = 0; ci < K1; ci++)
                a = fmaf(gs[(t3 + k) * K1 + ci], w3[((size_t)k * K1 + ci) * K2 + c], a);
        o[t3] = a;
    }
    float sc = sc3[c], sh = sh3[c];
    float p0 = (fmaf(sc, o[0], sh) + fmaf(sc, o[1], sh) + fmaf(sc, o[2], sh) + fmaf(sc, o[3], sh)) * 0.25f;
    float p1 = (fmaf(sc, o[4], sh) + fmaf(sc, o[5], sh) + fmaf(sc, o[6], sh) + fmaf(sc, o[7], sh)) * 0.25f;
    flat[c]      = fmaxf(p0, 0.f);
    flat[K2 + c] = fmaxf(p1, 0.f);
    __syncthreads();
    if (c < 4) {
        float a = db[c];
        for (int i = 0; i < 2 * K2; i++) a = fmaf(flat[i], dw[i * 4 + c], a);
        logits[c] = a;
    }
    __syncthreads();
    if (c == 0) {
        float mx = fmaxf(fmaxf(logits[0], logits[1]), fmaxf(logits[2], logits[3]));
        float s = expf(logits[0] - mx) + expf(logits[1] - mx) +
                  expf(logits[2] - mx) + expf(logits[3] - mx);
        float lse = logf(s);
        for (int d = 0; d < 4; d++) out[b * 4 + d] = logits[d] - mx - lse;
    }
}

// =====================================================================
extern "C" void kernel_launch(void* const* d_in, const int* in_sizes, int n_in,
                              void* d_out, int out_size, void* d_ws, size_t ws_size,
                              hipStream_t stream) {
    const float* x        = (const float*)d_in[0];
    const int*   eidx     = (const int*)d_in[1];
    const float* conv1_w  = (const float*)d_in[3];
    const float* bn1_g    = (const float*)d_in[4];
    const float* bn1_b    = (const float*)d_in[5];
    const float* bn1_m    = (const float*)d_in[6];
    const float* bn1_v    = (const float*)d_in[7];
    const float* gcn1_w   = (const float*)d_in[8];
    const float* gcn1_b   = (const float*)d_in[9];
    const float* gcn2_w   = (const float*)d_in[10];
    const float* gcn2_b   = (const float*)d_in[11];
    const float* conv2_w  = (const float*)d_in[12];
    const float* bn2_g    = (const float*)d_in[13];
    const float* bn2_b    = (const float*)d_in[14];
    const float* bn2_m    = (const float*)d_in[15];
    const float* bn2_v    = (const float*)d_in[16];
    const float* conv3_w  = (const float*)d_in[17];
    const float* bn3_g    = (const float*)d_in[18];
    const float* bn3_b    = (const float*)d_in[19];
    const float* bn3_m    = (const float*)d_in[20];
    const float* bn3_v    = (const float*)d_in[21];
    const float* dense_w  = (const float*)d_in[22];
    const float* dense_b  = (const float*)d_in[23];
    float* outp = (float*)d_out;

    char* ws = (char*)d_ws;
    float*    deg   = (float*)(ws + OFF_DEG);
    float*    norm  = (float*)(ws + OFF_NORM);
    float*    sc1   = (float*)(ws + OFF_SC1);
    float*    sh1   = (float*)(ws + OFF_SH1);
    float*    sc2   = (float*)(ws + OFF_SC2);
    float*    sh2   = (float*)(ws + OFF_SH2);
    float*    sc3   = (float*)(ws + OFF_SC3);
    float*    sh3   = (float*)(ws + OFF_SH3);
    uint32_t* wpk1  = (uint32_t*)(ws + OFF_WPK1);
    uint32_t* wpk2  = (uint32_t*)(ws + OFF_WPK2);
    uint32_t* wpk2c = (uint32_t*)(ws + OFF_WPK2C);
    uint16_t* h0    = (uint16_t*)(ws + OFF_H0);
    uint16_t* Hbuf  = (uint16_t*)(ws + OFF_H);
    uint16_t* Ybuf  = (uint16_t*)(ws + OFF_Y);
    float*    ACC   = (float*)(ws + OFF_ACC);
    float*    P     = (float*)(ws + OFF_P);
    float*    G     = (float*)(ws + OFF_G);

    const int* src = eidx;          // edge_index[0]
    const int* dst = eidx + EE;     // edge_index[1]

    // setup: BN constants + weight packing + graph norm
    k_prep<<<1, 64, 0, stream>>>(bn1_g, bn1_b, bn1_m, bn1_v, bn2_g, bn2_b, bn2_m, bn2_v,
                                 bn3_g, bn3_b, bn3_m, bn3_v, sc1, sh1, sc2, sh2, sc3, sh3);
    k_packW<<<1, 256, 0, stream>>>(gcn1_w, wpk1, 1, 1, 8, 128, K0 * EMB);        // 256 th
    k_packW<<<4, 256, 0, stream>>>(gcn2_w, wpk2, 1, 4, 8, 128, EMB * EMB);       // 1024 th
    k_packW<<<14, 256, 0, stream>>>(conv2_w, wpk2c, 7, 4, 4, 64, EMB * K1);      // 3584 th
    k_deginit<<<NN / 256, 256, 0, stream>>>(deg);
    k_degacc<<<EE / 256, 256, 0, stream>>>(dst, deg);
    k_norm<<<EE / 256, 256, 0, stream>>>(src, dst, deg, norm);

    // conv1 -> bf16 (N,T,32)
    k_conv1<<<MROWS / 256, 256, 0, stream>>>(x, conv1_w, sc1, sh1, h0);

    // GCN layer 1: GEMM (K=32) -> scatter -> bias+relu
    k_gemm<1><<<MROWS / 128, 256, 0, stream>>>(h0, wpk1, Ybuf);
    k_accinit<<<(MROWS * 16) / 256, 256, 0, stream>>>(Ybuf, deg, ACC);
    k_scatter<<<(EE * TT * 16) / 256, 256, 0, stream>>>(Ybuf, src, dst, norm, ACC);
    k_biasrelu<<<(MROWS * 16) / 256, 256, 0, stream>>>(ACC, gcn1_b, Hbuf);

    // GCN layer 2: GEMM (K=128) -> scatter -> bias+relu
    k_gemm<4><<<MROWS / 128, 256, 0, stream>>>(Hbuf, wpk2, Ybuf);
    k_accinit<<<(MROWS * 16) / 256, 256, 0, stream>>>(Ybuf, deg, ACC);
    k_scatter<<<(EE * TT * 16) / 256, 256, 0, stream>>>(Ybuf, src, dst, norm, ACC);
    k_biasrelu<<<(MROWS * 16) / 256, 256, 0, stream>>>(ACC, gcn2_b, Hbuf);

    // conv2 (WMMA tap-GEMMs) + bn2 + pool + relu
    k_conv2<<<NN, 256, 0, stream>>>(Hbuf, wpk2c, sc2, sh2, P);

    // per-graph mean pool + head
    k_graphpool<<<(BB * 14 * K1 + 255) / 256, 256, 0, stream>>>(P, G);
    k_head<<<BB, 64, 0, stream>>>(G, conv3_w, sc3, sh3, dense_w, dense_b, outp);

    (void)in_sizes; (void)n_in; (void)out_size; (void)ws_size;
}
